// SimpleDCTSGCNLayer_24180665876676
// MI455X (gfx1250) — matrix operand order repaired and checked by reading, not compiled, not run on hardware
//
#include <hip/hip_runtime.h>
#include <hip/hip_bf16.h>

typedef __bf16 bf16_t;
typedef __bf16 v16bf __attribute__((ext_vector_type(16)));
typedef __bf16 v8bf  __attribute__((ext_vector_type(8)));
typedef float  v8f   __attribute__((ext_vector_type(8)));
typedef float  v4f   __attribute__((ext_vector_type(4)));

#define DH 128
#define NEG_SLOPE 0.01f

// ---------------- utility kernels ----------------

// n must be a multiple of 4 (always true here)
__global__ void fill_zero_f32v4(float* __restrict__ p, long long n4) {
    long long i = (long long)blockIdx.x * blockDim.x + threadIdx.x;
    if (i < n4) {
        v4f z;
#pragma unroll
        for (int j = 0; j < 4; ++j) z[j] = 0.0f;
        ((v4f*)p)[i] = z;
    }
}

// W is [Kdim][Ndim] row-major f32; write Wt as [Ndim][Kdim] bf16 (transposed)
__global__ void convert_weight_t(const float* __restrict__ W, bf16_t* __restrict__ Wt,
                                 int Kdim, int Ndim) {
    int tid = blockIdx.x * blockDim.x + threadIdx.x;
    if (tid >= Kdim * Ndim) return;
    int k = tid / Ndim, n = tid % Ndim;
    Wt[(long long)n * Kdim + k] = (bf16_t)W[(long long)k * Ndim + n];
}

// gather x[src] rows (float4 per lane; one wave covers a full 128-col row),
// atomically accumulate at acc[dst], count in-degree once per edge.
__global__ void scatter_edges(const float* __restrict__ x,
                              const int* __restrict__ src,
                              const int* __restrict__ dst,
                              int nE,
                              float* __restrict__ acc,
                              float* __restrict__ deg) {
    long long tid = (long long)blockIdx.x * blockDim.x + threadIdx.x;
    long long total = (long long)nE * 32;           // 32 lanes/edge, 4 floats/lane
    if (tid >= total) return;
    int e = (int)(tid >> 5);
    int c = (int)(tid & 31) * 4;
    int s = src[e], d = dst[e];
    v4f v = *(const v4f*)(x + (long long)s * DH + c);
    float* a = acc + (long long)d * DH + c;
    atomicAdd(a + 0, v[0]);
    atomicAdd(a + 1, v[1]);
    atomicAdd(a + 2, v[2]);
    atomicAdd(a + 3, v[3]);
    if ((tid & 31) == 0) atomicAdd(deg + d, 1.0f);
}

// ---------------- WMMA GEMM stage 1 ----------------
// Cat[:, colOff:colOff+128] = bf16( leakyrelu( (acc/deg) @ W + b ) )
// A: f32 [M][Kdim] with per-row 1/max(deg,1) folded into the tile load.
// Wt: bf16 [128][Kdim] (transposed). One wave -> 16 rows x 128 cols.
__global__ __launch_bounds__(128)
void gemm_norm_wmma(const float* __restrict__ Amat, const float* __restrict__ deg,
                    const bf16_t* __restrict__ Wt, const float* __restrict__ bias,
                    bf16_t* __restrict__ Cat, int M, int Kdim, int colOff) {
    const int wave = threadIdx.x >> 5;
    const int lane = threadIdx.x & 31;
    const int rowTile = blockIdx.x * 4 + wave;
    if (rowTile * 16 >= M) return;
    const int row0 = rowTile * 16;
    const int m    = lane & 15;
    const int half = lane >> 4;

    const float rdeg = 1.0f / fmaxf(deg[row0 + m], 1.0f);

    v8f zero;
#pragma unroll
    for (int j = 0; j < 8; ++j) zero[j] = 0.0f;
    v8f accv[8];
#pragma unroll
    for (int t = 0; t < 8; ++t) accv[t] = zero;

    const float* arow = Amat + (long long)(row0 + m) * Kdim;
    const int kTiles = Kdim >> 5;
    for (int kt = 0; kt < kTiles; ++kt) {
        const float* ak = arow + kt * 32;
        // A 16x32 layout (matches B): K = 16*half + e
        v8f alo = *(const v8f*)(ak + 16 * half);
        v8f ahi = *(const v8f*)(ak + 16 * half + 8);
        v16bf afrag;
#pragma unroll
        for (int e = 0; e < 8; ++e) {
            afrag[e]     = (bf16_t)(alo[e] * rdeg);
            afrag[e + 8] = (bf16_t)(ahi[e] * rdeg);
        }
#pragma unroll
        for (int nt = 0; nt < 8; ++nt) {
            // B 32x16 layout: N = lane&15, K = 32*kt + 16*half + e (contiguous in Wt[N][K])
            const bf16_t* bp = Wt + (long long)(nt * 16 + m) * Kdim + kt * 32 + 16 * half;
            v16bf bfrag = *(const v16bf*)bp;
            accv[nt] = __builtin_amdgcn_wmma_f32_16x16x32_bf16(
                false, afrag, false, bfrag, (short)0, accv[nt], false, false);
        }
    }

#pragma unroll
    for (int nt = 0; nt < 8; ++nt) {
        const int col = nt * 16 + m;
        const float bv = bias[col];
#pragma unroll
        for (int r = 0; r < 8; ++r) {
            float v = accv[nt][r] + bv;
            v = (v > 0.0f) ? v : NEG_SLOPE * v;
            const int row = row0 + r + 8 * half;          // D: VGPR r -> M = r + 8*half
            Cat[(long long)row * 256 + colOff + col] = (bf16_t)v;
        }
    }
}

// ---------------- WMMA GEMM stage 2 ----------------
// Out = f32( Cat(bf16 [M][256]) @ Wc + bc ), no activation.
__global__ __launch_bounds__(128)
void gemm_out_wmma(const bf16_t* __restrict__ Cat, const bf16_t* __restrict__ Wt,
                   const float* __restrict__ bias, float* __restrict__ Out, int M) {
    const int Kdim = 256;
    const int wave = threadIdx.x >> 5;
    const int lane = threadIdx.x & 31;
    const int rowTile = blockIdx.x * 4 + wave;
    if (rowTile * 16 >= M) return;
    const int row0 = rowTile * 16;
    const int m    = lane & 15;
    const int half = lane >> 4;

    v8f zero;
#pragma unroll
    for (int j = 0; j < 8; ++j) zero[j] = 0.0f;
    v8f accv[8];
#pragma unroll
    for (int t = 0; t < 8; ++t) accv[t] = zero;

    const bf16_t* arow = Cat + (long long)(row0 + m) * Kdim;
    for (int kt = 0; kt < (Kdim >> 5); ++kt) {
        const bf16_t* ak = arow + kt * 32;
        v8bf alo = *(const v8bf*)(ak + 16 * half);
        v8bf ahi = *(const v8bf*)(ak + 16 * half + 8);
        v16bf afrag;
#pragma unroll
        for (int e = 0; e < 8; ++e) { afrag[e] = alo[e]; afrag[e + 8] = ahi[e]; }
#pragma unroll
        for (int nt = 0; nt < 8; ++nt) {
            const bf16_t* bp = Wt + (long long)(nt * 16 + m) * Kdim + kt * 32 + 16 * half;
            v16bf bfrag = *(const v16bf*)bp;
            accv[nt] = __builtin_amdgcn_wmma_f32_16x16x32_bf16(
                false, afrag, false, bfrag, (short)0, accv[nt], false, false);
        }
    }

#pragma unroll
    for (int nt = 0; nt < 8; ++nt) {
        const int col = nt * 16 + m;
        const float bv = bias[col];
#pragma unroll
        for (int r = 0; r < 8; ++r) {
            const int row = row0 + r + 8 * half;
            Out[(long long)row * DH + col] = accv[nt][r] + bv;
        }
    }
}

// ---------------- host launch ----------------

static inline size_t align256(size_t x) { return (x + 255) & ~(size_t)255; }

extern "C" void kernel_launch(void* const* d_in, const int* in_sizes, int n_in,
                              void* d_out, int out_size, void* d_ws, size_t ws_size,
                              hipStream_t stream) {
    const int NP = 300000, NS = 10000, D = DH;
    const float* x_paper = (const float*)d_in[0];
    const float* x_snap  = (const float*)d_in[1];
    const int* src_pf = (const int*)d_in[2];
    const int* dst_pf = (const int*)d_in[3];
    const int* src_pb = (const int*)d_in[4];
    const int* dst_pb = (const int*)d_in[5];
    const int* src_sf = (const int*)d_in[6];
    const int* dst_sf = (const int*)d_in[7];
    const int* src_sb = (const int*)d_in[8];
    const int* dst_sb = (const int*)d_in[9];
    const float* Wp_f = (const float*)d_in[10];
    const float* bp_f = (const float*)d_in[11];
    const float* Ws_f = (const float*)d_in[12];
    const float* bs_f = (const float*)d_in[13];
    const float* Wp_b = (const float*)d_in[14];
    const float* bp_b = (const float*)d_in[15];
    const float* Ws_b = (const float*)d_in[16];
    const float* bs_b = (const float*)d_in[17];
    const float* Wc_p = (const float*)d_in[18];
    const float* bc_p = (const float*)d_in[19];
    const float* Wc_s = (const float*)d_in[20];
    const float* bc_s = (const float*)d_in[21];

    const int EP = in_sizes[2];
    const int ES = in_sizes[6];
    const int KL = in_sizes[10] / (D * D);   // = 2 layers

    float* out_p = (float*)d_out;
    float* out_s = out_p + (long long)NP * D;

    // workspace carve
    char* w = (char*)d_ws;
    float* acc_p = (float*)w;  w += align256((size_t)NP * D * 4);
    float* deg_p = (float*)w;  w += align256((size_t)NP * 4);
    float* acc_s = (float*)w;  w += align256((size_t)NS * D * 4);
    float* deg_s = (float*)w;  w += align256((size_t)NS * 4);
    bf16_t* cat_p = (bf16_t*)w; w += align256((size_t)NP * 256 * 2);
    bf16_t* cat_s = (bf16_t*)w; w += align256((size_t)NS * 256 * 2);
    bf16_t* wt_pf = (bf16_t*)w; w += align256((size_t)D * D * 2);
    bf16_t* wt_pb = (bf16_t*)w; w += align256((size_t)D * D * 2);
    bf16_t* wt_sf = (bf16_t*)w; w += align256((size_t)D * D * 2);
    bf16_t* wt_sb = (bf16_t*)w; w += align256((size_t)D * D * 2);
    bf16_t* wt_cp = (bf16_t*)w; w += align256((size_t)2 * D * D * 2);
    bf16_t* wt_cs = (bf16_t*)w; w += align256((size_t)2 * D * D * 2);
    (void)ws_size; (void)n_in; (void)out_size;

    const int TPB = 256;
    const long long accP4 = (long long)NP * D / 4, accS4 = (long long)NS * D / 4;
    const long long degP4 = NP / 4, degS4 = NS / 4;
    const int gAccP = (int)((accP4 + TPB - 1) / TPB);
    const int gAccS = (int)((accS4 + TPB - 1) / TPB);
    const int gDegP = (int)((degP4 + TPB - 1) / TPB);
    const int gDegS = (int)((degS4 + TPB - 1) / TPB);
    const int gScP  = (int)(((long long)EP * 32 + TPB - 1) / TPB);
    const int gScS  = (int)(((long long)ES * 32 + TPB - 1) / TPB);
    const int gGemP = (NP / 16 + 3) / 4;   // 4 waves (16-row tiles) per 128-thread block
    const int gGemS = (NS / 16 + 3) / 4;
    const int gW1   = (D * D + TPB - 1) / TPB;
    const int gW2   = (2 * D * D + TPB - 1) / TPB;

    for (int i = 0; i < KL; ++i) {
        const float* hp = (i == 0) ? x_paper : out_p;
        const float* hs = (i == 0) ? x_snap  : out_s;

        // per-layer weights -> bf16, transposed [N][K]
        convert_weight_t<<<gW1, TPB, 0, stream>>>(Wp_f + (long long)i * D * D, wt_pf, D, D);
        convert_weight_t<<<gW1, TPB, 0, stream>>>(Wp_b + (long long)i * D * D, wt_pb, D, D);
        convert_weight_t<<<gW1, TPB, 0, stream>>>(Ws_f + (long long)i * D * D, wt_sf, D, D);
        convert_weight_t<<<gW1, TPB, 0, stream>>>(Ws_b + (long long)i * D * D, wt_sb, D, D);
        convert_weight_t<<<gW2, TPB, 0, stream>>>(Wc_p + (long long)i * 2 * D * D, wt_cp, 2 * D, D);
        convert_weight_t<<<gW2, TPB, 0, stream>>>(Wc_s + (long long)i * 2 * D * D, wt_cs, 2 * D, D);

        // ---- paper forward ----
        fill_zero_f32v4<<<gAccP, TPB, 0, stream>>>(acc_p, accP4);
        fill_zero_f32v4<<<gDegP, TPB, 0, stream>>>(deg_p, degP4);
        scatter_edges<<<gScP, TPB, 0, stream>>>(hp, src_pf, dst_pf, EP, acc_p, deg_p);
        gemm_norm_wmma<<<gGemP, 128, 0, stream>>>(acc_p, deg_p, wt_pf, bp_f + (long long)i * D,
                                                  cat_p, NP, D, 0);
        // ---- paper backward (reuse acc/deg) ----
        fill_zero_f32v4<<<gAccP, TPB, 0, stream>>>(acc_p, accP4);
        fill_zero_f32v4<<<gDegP, TPB, 0, stream>>>(deg_p, degP4);
        scatter_edges<<<gScP, TPB, 0, stream>>>(hp, src_pb, dst_pb, EP, acc_p, deg_p);
        gemm_norm_wmma<<<gGemP, 128, 0, stream>>>(acc_p, deg_p, wt_pb, bp_b + (long long)i * D,
                                                  cat_p, NP, D, D);
        // ---- snap forward ----
        fill_zero_f32v4<<<gAccS, TPB, 0, stream>>>(acc_s, accS4);
        fill_zero_f32v4<<<gDegS, TPB, 0, stream>>>(deg_s, degS4);
        scatter_edges<<<gScS, TPB, 0, stream>>>(hs, src_sf, dst_sf, ES, acc_s, deg_s);
        gemm_norm_wmma<<<gGemS, 128, 0, stream>>>(acc_s, deg_s, wt_sf, bs_f + (long long)i * D,
                                                  cat_s, NS, D, 0);
        // ---- snap backward ----
        fill_zero_f32v4<<<gAccS, TPB, 0, stream>>>(acc_s, accS4);
        fill_zero_f32v4<<<gDegS, TPB, 0, stream>>>(deg_s, degS4);
        scatter_edges<<<gScS, TPB, 0, stream>>>(hs, src_sb, dst_sb, ES, acc_s, deg_s);
        gemm_norm_wmma<<<gGemS, 128, 0, stream>>>(acc_s, deg_s, wt_sb, bs_b + (long long)i * D,
                                                  cat_s, NS, D, D);

        // ---- concat linear: h = cat @ Wc + bc  (writes d_out, read next iter) ----
        gemm_out_wmma<<<gGemP, 128, 0, stream>>>(cat_p, wt_cp, bc_p + (long long)i * D, out_p, NP);
        gemm_out_wmma<<<gGemS, 128, 0, stream>>>(cat_s, wt_cs, bc_s + (long long)i * D, out_s, NS);
    }
}
